// GATLayer_sparse_20401094656457
// MI455X (gfx1250) — matrix operand order repaired and checked by reading, not compiled, not run on hardware
//
#include <hip/hip_runtime.h>
#include <hip/hip_bf16.h>

typedef __attribute__((ext_vector_type(16))) _Float16 v16h;
typedef __attribute__((ext_vector_type(8)))  _Float16 v8h;
typedef __attribute__((ext_vector_type(8)))  float    v8f;

#define GAT_N     50000
#define GAT_E     800000
#define GAT_FIN   256
#define GAT_FOUT  128
#define GAT_ALPHA 0.2f

// ---------------- workspace layout (bytes, 64-aligned) ----------------
// xh   : N*FIN  f16  = 25,600,000
// wt   : FOUT*FIN f16 (transposed W) = 65,536
// h    : N*FOUT f32  = 25,600,000
// ssrc : N f32
// sdst : N f32
// mmax : N u32 (monotonic-mapped float max)
// denom: N f32
// exb  : E f32
#define OFF_XH    ((size_t)0)
#define OFF_WT    ((size_t)25600000)
#define OFF_H     ((size_t)25665536)
#define OFF_SSRC  ((size_t)51265536)
#define OFF_SDST  ((size_t)51465536)
#define OFF_MMAX  ((size_t)51665536)
#define OFF_DENOM ((size_t)51865536)
#define OFF_EXB   ((size_t)52065536)

// ---------------- K0: zero/init ----------------
__global__ void k_init(float* __restrict__ out, unsigned* __restrict__ mmax,
                       float* __restrict__ denom) {
  size_t i = (size_t)blockIdx.x * blockDim.x + threadIdx.x;
  size_t total = (size_t)GAT_N * GAT_FOUT;
  for (; i < total; i += (size_t)gridDim.x * blockDim.x) {
    out[i] = 0.0f;
    if (i < GAT_N) { mmax[i] = 0u; denom[i] = 0.0f; }
  }
}

// ---------------- K1: convert x -> f16, W -> Wt f16 ----------------
__global__ void k_convert(const float* __restrict__ x, const float* __restrict__ W,
                          _Float16* __restrict__ xh, _Float16* __restrict__ wt) {
  size_t i = (size_t)blockIdx.x * blockDim.x + threadIdx.x;
  size_t nx = (size_t)GAT_N * GAT_FIN;
  for (; i < nx; i += (size_t)gridDim.x * blockDim.x) {
    xh[i] = (_Float16)x[i];
    if (i < (size_t)GAT_FIN * GAT_FOUT) {
      int k = (int)(i / GAT_FOUT);
      int c = (int)(i % GAT_FOUT);
      wt[(size_t)c * GAT_FIN + k] = (_Float16)W[i];  // Wt[c][k] = W[k][c]
    }
  }
}

// ---------------- K2: h = x @ W via WMMA f32_16x16x32_f16 ----------------
// grid = 3125 blocks (one 16-row stripe), 256 threads = 8 waves (one 16-col tile each)
__global__ void k_gemm_wmma(const _Float16* __restrict__ xh,
                            const _Float16* __restrict__ wt,
                            float* __restrict__ h) {
  const int wave = threadIdx.x >> 5;       // 0..7 -> col tile
  const int lane = threadIdx.x & 31;
  const int mn   = lane & 15;              // M for A, N for B/C
  const int hs   = lane >> 4;              // lane-half select
  const int row0 = blockIdx.x * 16;
  const int col0 = wave * 16;

  const _Float16* arow = xh + (size_t)(row0 + mn) * GAT_FIN;
  const _Float16* bcol = wt + (size_t)(col0 + mn) * GAT_FIN;

  v8f c = {};
#pragma unroll
  for (int k0 = 0; k0 < GAT_FIN; k0 += 32) {
    // A fragment (16x32 f16): VGPR0-3 hold K = hs*8 + 0..7, VGPR4-7 hold K = 16 + hs*8 + 0..7
    v16h a;
    ((v8h*)&a)[0] = *(const v8h*)(arow + k0 + hs * 8);
    ((v8h*)&a)[1] = *(const v8h*)(arow + k0 + 16 + hs * 8);
    // B fragment (32x16 f16): VGPR0-3 hold K = hs*8 + 0..7, VGPR4-7 hold K = 16 + hs*8 + 0..7 for column N=mn
    v16h b;
    ((v8h*)&b)[0] = *(const v8h*)(bcol + k0 + hs * 8);
    ((v8h*)&b)[1] = *(const v8h*)(bcol + k0 + 16 + hs * 8);
    c = __builtin_amdgcn_wmma_f32_16x16x32_f16(false, a, false, b,
                                               (short)0, c, false, false);
  }
  // C/D layout: VGPR r -> (M = r + hs*8, N = mn)
#pragma unroll
  for (int r = 0; r < 8; ++r) {
    h[(size_t)(row0 + hs * 8 + r) * GAT_FOUT + col0 + mn] = c[r];
  }
}

// ---------------- K3: s_src / s_dst (one wave per row) ----------------
__global__ void k_scores(const float* __restrict__ h, const float* __restrict__ a,
                         float* __restrict__ ssrc, float* __restrict__ sdst) {
  int wave = threadIdx.x >> 5;
  int lane = threadIdx.x & 31;
  int row  = blockIdx.x * 8 + wave;
  if (row >= GAT_N) return;
  const float* hr = h + (size_t)row * GAT_FOUT;
  float a1 = 0.0f, a2 = 0.0f;
#pragma unroll
  for (int q = 0; q < 4; ++q) {
    int f = lane + 32 * q;
    float hv = hr[f];
    a1 += hv * a[f];
    a2 += hv * a[GAT_FOUT + f];
  }
#pragma unroll
  for (int off = 16; off > 0; off >>= 1) {
    a1 += __shfl_xor(a1, off, 32);
    a2 += __shfl_xor(a2, off, 32);
  }
  if (lane == 0) { ssrc[row] = a1; sdst[row] = a2; }
}

__device__ __forceinline__ float leaky(float t) {
  return t > 0.0f ? t : GAT_ALPHA * t;
}
__device__ __forceinline__ unsigned fmap(float f) {
  unsigned u = __float_as_uint(f);
  return (u & 0x80000000u) ? ~u : (u | 0x80000000u);
}
__device__ __forceinline__ float funmap(unsigned m) {
  return __uint_as_float((m & 0x80000000u) ? (m ^ 0x80000000u) : ~m);
}

// ---------------- K4: segment max of e over src ----------------
__global__ void k_edge_max(const int* __restrict__ edge,
                           const float* __restrict__ ssrc, const float* __restrict__ sdst,
                           unsigned* __restrict__ mmax) {
  int e = blockIdx.x * blockDim.x + threadIdx.x;
  if (e >= GAT_E) return;
  int s = edge[e], d = edge[GAT_E + e];
  float t = leaky(ssrc[s] + sdst[d]);
  atomicMax(&mmax[s], fmap(t));
}

// ---------------- K5: ex = exp(e - m[src]); denom += ex ----------------
__global__ void k_edge_exp(const int* __restrict__ edge,
                           const float* __restrict__ ssrc, const float* __restrict__ sdst,
                           const unsigned* __restrict__ mmax,
                           float* __restrict__ denom, float* __restrict__ exb) {
  int e = blockIdx.x * blockDim.x + threadIdx.x;
  if (e >= GAT_E) return;
  int s = edge[e], d = edge[GAT_E + e];
  float t  = leaky(ssrc[s] + sdst[d]);
  float ex = __expf(t - funmap(mmax[s]));
  exb[e] = ex;
  atomicAdd(&denom[s], ex);
}

// ---------------- K6: out[src] += (ex/denom[src]) * h[dst] (wave per edge) ----------------
__global__ void k_aggregate(const int* __restrict__ edge,
                            const float* __restrict__ h,
                            const float* __restrict__ exb,
                            const float* __restrict__ denom,
                            float* __restrict__ out) {
  int e    = (int)(((size_t)blockIdx.x * blockDim.x + threadIdx.x) >> 5);
  int lane = threadIdx.x & 31;
  if (e >= GAT_E) return;
  int s = edge[e], d = edge[GAT_E + e];
  float w = exb[e] / denom[s];
  float4 v = ((const float4*)(h + (size_t)d * GAT_FOUT))[lane];
  float* o = out + (size_t)s * GAT_FOUT + lane * 4;
  atomicAdd(o + 0, w * v.x);
  atomicAdd(o + 1, w * v.y);
  atomicAdd(o + 2, w * v.z);
  atomicAdd(o + 3, w * v.w);
}

// ---------------- K7: ELU in place ----------------
__global__ void k_elu(float* __restrict__ out) {
  size_t i = (size_t)blockIdx.x * blockDim.x + threadIdx.x;
  size_t total = (size_t)GAT_N * GAT_FOUT;
  for (; i < total; i += (size_t)gridDim.x * blockDim.x) {
    float v = out[i];
    out[i] = v > 0.0f ? v : (__expf(v) - 1.0f);
  }
}

extern "C" void kernel_launch(void* const* d_in, const int* in_sizes, int n_in,
                              void* d_out, int out_size, void* d_ws, size_t ws_size,
                              hipStream_t stream) {
  const float* x    = (const float*)d_in[0];
  const float* W    = (const float*)d_in[1];
  const float* a    = (const float*)d_in[2];
  const int*   edge = (const int*)d_in[3];

  char* ws = (char*)d_ws;
  _Float16* xh    = (_Float16*)(ws + OFF_XH);
  _Float16* wt    = (_Float16*)(ws + OFF_WT);
  float*    h     = (float*)(ws + OFF_H);
  float*    ssrc  = (float*)(ws + OFF_SSRC);
  float*    sdst  = (float*)(ws + OFF_SDST);
  unsigned* mmax  = (unsigned*)(ws + OFF_MMAX);
  float*    denom = (float*)(ws + OFF_DENOM);
  float*    exb   = (float*)(ws + OFF_EXB);
  float*    out   = (float*)d_out;

  // K0: zero output accumulator + init segment buffers
  k_init<<<2048, 256, 0, stream>>>(out, mmax, denom);
  // K1: precision conversion / transpose
  k_convert<<<2048, 256, 0, stream>>>(x, W, xh, wt);
  // K2: WMMA GEMM: 3125 row tiles x (8 waves = 8 col tiles)
  k_gemm_wmma<<<GAT_N / 16, 256, 0, stream>>>(xh, wt, h);
  // K3: attention score halves, one wave32 per row
  k_scores<<<(GAT_N + 7) / 8, 256, 0, stream>>>(h, a, ssrc, sdst);
  // K4: segment max
  k_edge_max<<<(GAT_E + 255) / 256, 256, 0, stream>>>(edge, ssrc, sdst, mmax);
  // K5: exp + segment sum
  k_edge_exp<<<(GAT_E + 255) / 256, 256, 0, stream>>>(edge, ssrc, sdst, mmax, denom, exb);
  // K6: weighted scatter-aggregate, one wave32 per edge
  k_aggregate<<<GAT_E / 8, 256, 0, stream>>>(edge, h, exb, denom, out);
  // K7: ELU
  k_elu<<<2048, 256, 0, stream>>>(out);
}
